// Net_54185307406400
// MI455X (gfx1250) — matrix operand order repaired and checked by reading, compiled, not run on hardware
//
#include <hip/hip_runtime.h>
#include <hip/hip_bf16.h>

#define B_SZ 512
#define SEQ_L 28
#define IN_F 28
#define D_MODEL 128
#define D_STATE 16
#define D_INNER 256
#define DT_RANK 8
#define N_CLS 10
#define M_ROWS (B_SZ * SEQ_L)              /* 14336 */
#define XZ_N (2 * D_INNER)                 /* 512   */
#define DBL_N (DT_RANK + 2 * D_STATE)      /* 40    */

typedef _Float16 half8 __attribute__((ext_vector_type(8)));
typedef _Float16 v16h  __attribute__((ext_vector_type(16)));
typedef float    v8f   __attribute__((ext_vector_type(8)));

// ---------------------------------------------------------------------------
// Prep: transpose f32 weights to N-major f16 so WMMA B-fragments are
// contiguous 16-half loads per lane (k = 16*(lane/16) + s, same order as the A fragment)
// ---------------------------------------------------------------------------
__global__ void prep_weights(const float* __restrict__ w_in,
                             const float* __restrict__ w_out,
                             _Float16* __restrict__ w_inT,
                             _Float16* __restrict__ w_outT) {
    int idx = blockIdx.x * blockDim.x + threadIdx.x;
    if (idx < D_MODEL * XZ_N) {            // w_inT[n][k] = w_in[k][n]  (512 x 128)
        int n = idx / D_MODEL, k = idx % D_MODEL;
        w_inT[idx] = (_Float16)w_in[k * XZ_N + n];
    }
    int j = idx - D_MODEL * XZ_N;
    if (j >= 0 && j < D_INNER * D_MODEL) { // w_outT[n][k] = w_out[k][n] (128 x 256)
        int n = j / D_INNER, k = j % D_INNER;
        w_outT[j] = (_Float16)w_out[k * D_MODEL + n];
    }
}

// ---------------------------------------------------------------------------
// Input projection: u = x @ w_inp  (K=28, too small for WMMA); emit f16 rows
// One block per output row; x row staged in LDS, w_inp reads coalesced.
// ---------------------------------------------------------------------------
__global__ void in_proj_kernel(const float* __restrict__ x,
                               const float* __restrict__ w_inp,
                               _Float16* __restrict__ u16) {
    __shared__ float xr[IN_F];
    int m = blockIdx.x;
    int d = threadIdx.x;
    if (d < IN_F) xr[d] = x[m * IN_F + d];
    __syncthreads();
    float s = 0.f;
#pragma unroll
    for (int k = 0; k < IN_F; ++k) s += xr[k] * w_inp[k * D_MODEL + d];
    u16[(size_t)m * D_MODEL + d] = (_Float16)s;
}

// ---------------------------------------------------------------------------
// WMMA GEMM: C[M,N] = A[M,K](f16,row-major) * BT[N,K](f16,N-major), f32 accum.
// Each wave32 computes 1 M-tile x NPW N-tiles. Per k-step: the A fragment and
// ALL NPW B fragments are loaded first (one batched load clause), then NPW
// v_wmma issue back-to-back, so load waits overlap WMMA execution instead of
// serializing load->wait->wmma per tile. A and B both use k = 16*(lane/16) + s order.
// ---------------------------------------------------------------------------
template <int K, int NTILES, int NPW>
__global__ __launch_bounds__(128) void wmma_gemm_f16(
        const _Float16* __restrict__ A, const _Float16* __restrict__ BT,
        float* __restrict__ C) {
    const int wave = threadIdx.x >> 5;
    const int lane = threadIdx.x & 31;
    const int NG   = NTILES / NPW;            // N-tile groups
    const int tile = blockIdx.x * 4 + wave;
    const int mt   = tile / NG;
    const int nt0  = (tile % NG) * NPW;
    const int r16  = lane & 15;
    const int hiA  = (lane < 16) ? 0 : 16;    // K sub-base for A fragment
    const int hiB  = (lane < 16) ? 0 : 16;    // K sub-base for B fragment
    const _Float16* ap = A  + (size_t)(mt * 16 + r16) * K;
    const _Float16* bp = BT + (size_t)(nt0 * 16 + r16) * K;

    v8f c[NPW];
#pragma unroll
    for (int j = 0; j < NPW; ++j) c[j] = (v8f){};

#pragma unroll
    for (int k0 = 0; k0 < K; k0 += 32) {
        // ---- batched fragment loads (A + all NPW B tiles) ----
        half8 lo = *(const half8*)(ap + k0 + hiA);
        half8 hi = *(const half8*)(ap + k0 + hiA + 8);
        v16h bfrag[NPW];
#pragma unroll
        for (int j = 0; j < NPW; ++j)
            bfrag[j] = *(const v16h*)(bp + (size_t)j * 16 * K + k0 + hiB);
        v16h a;
#pragma unroll
        for (int i = 0; i < 8; ++i) { a[i] = lo[i]; a[i + 8] = hi[i]; }
        // ---- WMMA burst ----
#pragma unroll
        for (int j = 0; j < NPW; ++j)
            c[j] = __builtin_amdgcn_wmma_f32_16x16x32_f16(
                       /*neg_a=*/false, a, /*neg_b=*/false, bfrag[j],
                       /*c_mod=*/(short)0, c[j],
                       /*reuse_a=*/false, /*reuse_b=*/false);
    }
    const int N = NTILES * 16;
    const int row0 = mt * 16 + ((lane < 16) ? 0 : 8);
    const int col0 = nt0 * 16 + r16;
#pragma unroll
    for (int j = 0; j < NPW; ++j)
#pragma unroll
        for (int r = 0; r < 8; ++r)
            C[(size_t)(row0 + r) * N + col0 + j * 16] = c[j][r];
}

__device__ __forceinline__ float silu_f(float v) {
    return v * (1.f / (1.f + __expf(-v)));
}

// ---------------------------------------------------------------------------
// Fused conv -> x-proj -> softplus(delta) -> selective scan -> gate.
// One workgroup per batch sample, one thread per d_inner channel.
// LDS: xc tile (28x256 f32) + dbl tile (28x40 f32)  ~33KB of 320KB WGP LDS.
// State h[16] lives in registers (fully unrolled).
// ---------------------------------------------------------------------------
__global__ __launch_bounds__(D_INNER) void scan_kernel(
        const float* __restrict__ xz,
        const float* __restrict__ conv_w, const float* __restrict__ conv_b,
        const float* __restrict__ w_x,    const float* __restrict__ w_dt,
        const float* __restrict__ dt_bias,const float* __restrict__ A_log,
        const float* __restrict__ Dvec,   _Float16* __restrict__ y16) {
    __shared__ float xc_lds[SEQ_L][D_INNER];
    __shared__ float dbl_lds[SEQ_L][DBL_N];
    const int b  = blockIdx.x;
    const int ch = threadIdx.x;
    const float* xzb = xz + (size_t)b * SEQ_L * XZ_N;

    // Depthwise causal conv (k=3) + SiLU, rolling window in registers
    const float w0 = conv_w[ch * 3 + 0], w1 = conv_w[ch * 3 + 1], w2 = conv_w[ch * 3 + 2];
    const float cb = conv_b[ch];
    float x0 = 0.f, x1 = 0.f;
    for (int t = 0; t < SEQ_L; ++t) {
        float x2 = xzb[t * XZ_N + ch];
        float v = x0 * w0 + x1 * w1 + x2 * w2 + cb;
        xc_lds[t][ch] = silu_f(v);
        x0 = x1; x1 = x2;
    }
    __syncthreads();

    // dbl = xc @ w_x  (28 x 40 outputs, K=256 over LDS)
    for (int idx = ch; idx < SEQ_L * DBL_N; idx += D_INNER) {
        int t = idx / DBL_N, j = idx % DBL_N;
        float s = 0.f;
        for (int k = 0; k < D_INNER; ++k) s += xc_lds[t][k] * w_x[k * DBL_N + j];
        dbl_lds[t][j] = s;
    }
    __syncthreads();

    float wdt[DT_RANK];
#pragma unroll
    for (int r = 0; r < DT_RANK; ++r) wdt[r] = w_dt[r * D_INNER + ch];
    const float dtb = dt_bias[ch];
    float Av[D_STATE];
#pragma unroll
    for (int s = 0; s < D_STATE; ++s) Av[s] = -__expf(A_log[ch * D_STATE + s]);
    const float Dch = Dvec[ch];

    float h[D_STATE];
#pragma unroll
    for (int s = 0; s < D_STATE; ++s) h[s] = 0.f;

    for (int t = 0; t < SEQ_L; ++t) {
        float acc = dtb;
#pragma unroll
        for (int r = 0; r < DT_RANK; ++r) acc += dbl_lds[t][r] * wdt[r];
        // softplus (numerically stable)
        float delta = fmaxf(acc, 0.f) + log1pf(__expf(-fabsf(acc)));
        float xc_t = xc_lds[t][ch];
        float y = 0.f;
#pragma unroll
        for (int s = 0; s < D_STATE; ++s) {
            float dA = __expf(delta * Av[s]);
            h[s] = dA * h[s] + (delta * dbl_lds[t][DT_RANK + s]) * xc_t;
            y += h[s] * dbl_lds[t][DT_RANK + D_STATE + s];
        }
        y += xc_t * Dch;
        float z = xzb[t * XZ_N + D_INNER + ch];
        y *= silu_f(z);
        y16[((size_t)b * SEQ_L + t) * D_INNER + ch] = (_Float16)y;
    }
}

// ---------------------------------------------------------------------------
// Mean-pool over L, then @ w_cls. One block per sample.
// ---------------------------------------------------------------------------
__global__ void head_kernel(const float* __restrict__ m,
                            const float* __restrict__ w_cls,
                            float* __restrict__ out) {
    __shared__ float pooled[D_MODEL];
    const int b = blockIdx.x;
    const int d = threadIdx.x;   // 128 threads
    float s = 0.f;
    for (int t = 0; t < SEQ_L; ++t) s += m[((size_t)b * SEQ_L + t) * D_MODEL + d];
    pooled[d] = s * (1.f / SEQ_L);
    __syncthreads();
    if (d < N_CLS) {
        float o = 0.f;
        for (int k = 0; k < D_MODEL; ++k) o += pooled[k] * w_cls[k * N_CLS + d];
        out[b * N_CLS + d] = o;
    }
}

// ---------------------------------------------------------------------------
extern "C" void kernel_launch(void* const* d_in, const int* in_sizes, int n_in,
                              void* d_out, int out_size, void* d_ws, size_t ws_size,
                              hipStream_t stream) {
    const float* x       = (const float*)d_in[0];
    const float* w_inp   = (const float*)d_in[1];
    const float* w_in    = (const float*)d_in[2];
    const float* conv_w  = (const float*)d_in[3];
    const float* conv_b  = (const float*)d_in[4];
    const float* w_x     = (const float*)d_in[5];
    const float* w_dt    = (const float*)d_in[6];
    const float* dt_bias = (const float*)d_in[7];
    const float* A_log   = (const float*)d_in[8];
    const float* Dv      = (const float*)d_in[9];
    const float* w_out   = (const float*)d_in[10];
    const float* w_cls   = (const float*)d_in[11];
    float* out = (float*)d_out;

    // workspace layout (256B-aligned offsets)
    char* ws = (char*)d_ws;
    _Float16* w_inT  = (_Float16*)(ws + 0);         //  512*128 f16 = 131072
    _Float16* w_outT = (_Float16*)(ws + 131072);    //  128*256 f16 =  65536
    _Float16* u16    = (_Float16*)(ws + 196608);    // 14336*128 f16 = 3670016
    float*    xz     = (float*)   (ws + 3866624);   // 14336*512 f32 = 29360128
    _Float16* y16    = (_Float16*)(ws + 33226752);  // 14336*256 f16 = 7340032
    float*    mbuf   = (float*)   (ws + 40566784);  // 14336*128 f32 = 7340032

    prep_weights<<<384, 256, 0, stream>>>(w_in, w_out, w_inT, w_outT);
    in_proj_kernel<<<M_ROWS, D_MODEL, 0, stream>>>(x, w_inp, u16);
    // xz = u @ w_in : 896 M-tiles x 32 N-tiles, K=128, 4 N-tiles per wave
    //   -> 896 * (32/4) = 7168 waves = 1792 blocks of 4 waves
    wmma_gemm_f16<D_MODEL, 32, 4><<<1792, 128, 0, stream>>>(u16, w_inT, xz);
    scan_kernel<<<B_SZ, D_INNER, 0, stream>>>(xz, conv_w, conv_b, w_x, w_dt,
                                              dt_bias, A_log, Dv, y16);
    // m = y @ w_out : 896 M-tiles x 8 N-tiles, K=256, 4 N-tiles per wave
    //   -> 896 * (8/4) = 1792 waves = 448 blocks of 4 waves
    wmma_gemm_f16<D_INNER, 8, 4><<<448, 128, 0, stream>>>(y16, w_outT, mbuf);
    head_kernel<<<B_SZ, D_MODEL, 0, stream>>>(mbuf, w_cls, out);
}
